// GAT_47485158425274
// MI455X (gfx1250) — matrix operand order repaired and checked by reading, not compiled, not run on hardware
//
#include <hip/hip_runtime.h>
#include <cstdint>

// ---------------------------------------------------------------------------
// GATv2 x3 + BN + fused regressor for MI455X (gfx1250, wave32, WMMA).
// Edge phase = memory/atomic bound (xl/xr live in 192MB L2); node transforms
// for layers 2/3 use v_wmma_f32_16x16x32_f16; regressor folded algebraically.
// ---------------------------------------------------------------------------

typedef __attribute__((ext_vector_type(16))) _Float16 v16h;
typedef __attribute__((ext_vector_type(8)))  float    v8f;

__device__ __forceinline__ float lrelu(float x, float s) { return x > 0.f ? x : s * x; }

// Monotone float<->uint encoding so segment-max uses native u32 atomic max.
__device__ __forceinline__ unsigned encF(float f) {
  unsigned u = __float_as_uint(f);
  return (u & 0x80000000u) ? ~u : (u | 0x80000000u);
}
__device__ __forceinline__ float decF(unsigned u) {
  return (u & 0x80000000u) ? __uint_as_float(u & 0x7FFFFFFFu) : __uint_as_float(~u);
}

// ------------------------------ utility ------------------------------------
__global__ void zero_f32(float* __restrict__ p, long n) {
  long i = (long)blockIdx.x * blockDim.x + threadIdx.x;
  if (i < n) p[i] = 0.f;
}

// Sum edge_attr columns (7) -> accum[0..6]
__global__ __launch_bounds__(256)
void ea_reduce(const float* __restrict__ ea, float* __restrict__ accum, int E_) {
  __shared__ float ls[7];
  int t = threadIdx.x;
  if (t < 7) ls[t] = 0.f;
  __syncthreads();
  float loc[7] = {0.f, 0.f, 0.f, 0.f, 0.f, 0.f, 0.f};
  for (long e = (long)blockIdx.x * blockDim.x + t; e < E_;
       e += (long)gridDim.x * blockDim.x) {
#pragma unroll
    for (int f = 0; f < 7; ++f) loc[f] += ea[e * 7 + f];
  }
#pragma unroll
  for (int f = 0; f < 7; ++f) atomicAdd(&ls[f], loc[f]);
  __syncthreads();
  if (t < 7) atomicAdd(&accum[t], ls[t]);
}

// 1 block, 32 threads: self-loop edge transforms + folded regressor weights.
__global__ void precompute(const float* __restrict__ accum,
                           const float* __restrict__ We1, const float* __restrict__ We2,
                           const float* __restrict__ Wreg, const float* __restrict__ breg,
                           const float* __restrict__ Wend, const float* __restrict__ bend,
                           float* __restrict__ eeself1, float* __restrict__ eeself2,
                           float* __restrict__ wcomb, float* __restrict__ c0,
                           int E_, int REG) {
  int c = threadIdx.x;  // 0..31
  float m[7];
#pragma unroll
  for (int f = 0; f < 7; ++f) m[f] = accum[f] / (float)E_;
  float s1 = 0.f, s2 = 0.f;
#pragma unroll
  for (int f = 0; f < 7; ++f) { s1 += m[f] * We1[f * 32 + c]; s2 += m[f] * We2[f * 32 + c]; }
  eeself1[c] = s1;
  eeself2[c] = s2;
  float wc = 0.f;
  for (int j = 0; j < REG; ++j) wc += Wreg[c * REG + j] * Wend[j];
  wcomb[c] = wc;
  if (c == 0) {
    float acc = bend[0];
    for (int j = 0; j < REG; ++j) acc += breg[j] * Wend[j];
    c0[0] = acc;
  }
}

// ----------------------- layer-1 node transform (Fin=2) --------------------
__global__ void transform_f2(const float* __restrict__ x,
                             const float* __restrict__ Wl, const float* __restrict__ bl,
                             const float* __restrict__ Wr, const float* __restrict__ br,
                             float* __restrict__ xl, float* __restrict__ xr, int n) {
  long i = (long)blockIdx.x * blockDim.x + threadIdx.x;
  if (i >= (long)n * 32) return;
  int nd = (int)(i >> 5), c = (int)(i & 31);
  float x0 = x[nd * 2], x1 = x[nd * 2 + 1];
  xl[i] = x0 * Wl[c] + x1 * Wl[32 + c] + bl[c];
  xr[i] = x0 * Wr[c] + x1 * Wr[32 + c] + br[c];
}

// ------------------ layers 2/3 node transform via WMMA ---------------------
// Computes [xl|xr] = xin(Nx32) @ [Wl|Wr](32x64) + [bl|br] in 16-node tiles.
// One wave per tile iteration; B tiles (weights) preloaded once per wave.
__global__ __launch_bounds__(128)
void transform_wmma(const float* __restrict__ xin,
                    const float* __restrict__ Wl, const float* __restrict__ bl,
                    const float* __restrict__ Wr, const float* __restrict__ br,
                    float* __restrict__ xl, float* __restrict__ xr, int n) {
  const int lane = threadIdx.x & 31;
  const int col  = lane & 15;
  const int half = lane >> 4;
  const int wave = blockIdx.x * (blockDim.x >> 5) + (threadIdx.x >> 5);
  const int nwaves = gridDim.x * (blockDim.x >> 5);

  // B (32x16), this code's k order, same as A: lanes 0-15 = cols, K rows [half*16, half*16+16)
  v16h B[4];
#pragma unroll
  for (int t = 0; t < 4; ++t) {
    const float* W = (t < 2) ? Wl : Wr;
    int cb = (t & 1) * 16 + col;
#pragma unroll
    for (int j = 0; j < 16; ++j) B[t][j] = (_Float16)W[(half * 16 + j) * 32 + cb];
  }
  float biasv[4];
  biasv[0] = bl[col]; biasv[1] = bl[16 + col];
  biasv[2] = br[col]; biasv[3] = br[16 + col];

  const int ntiles = n >> 4;  // N = 100000 = 6250*16 exactly
  for (int tile = wave; tile < ntiles; tile += nwaves) {
    const int row = tile * 16 + (lane & 15);
    const float* xrow = xin + (long)row * 32;
    // A (16x32), this code's k order, same as B: elements 2v,2v+1 hold k = half*16 + v*2, +1
    v16h A;
#pragma unroll
    for (int v = 0; v < 8; ++v) {
      int kb = half * 16 + v * 2;
      A[2 * v]     = (_Float16)xrow[kb];
      A[2 * v + 1] = (_Float16)xrow[kb + 1];
    }
#pragma unroll
    for (int t = 0; t < 4; ++t) {
      v8f Cz = {};
      v8f D = __builtin_amdgcn_wmma_f32_16x16x32_f16(
          /*neg_a=*/false, A, /*neg_b=*/false, B[t],
          /*c_mod=*/(short)0, Cz, /*reuse_a=*/false, /*reuse_b=*/false);
      float* outp = (t < 2) ? xl : xr;
      const int cb = (t & 1) * 16 + col;
#pragma unroll
      for (int r = 0; r < 8; ++r) {
        int m = half * 8 + r;  // C/D: vgpr r holds M=r (lanes<16) / M=r+8 (lanes>=16)
        outp[(long)(tile * 16 + m) * 32 + cb] = D[r] + biasv[t];
      }
    }
  }
}

// --------------------------- edge pass A: logits + max ---------------------
__global__ __launch_bounds__(256)
void edge_logits(const int* __restrict__ src, const int* __restrict__ dst,
                 const float* __restrict__ ea, const float* __restrict__ We,
                 const float* __restrict__ att, const float* __restrict__ eeself,
                 const float* __restrict__ xl, const float* __restrict__ xr,
                 float* __restrict__ logits, unsigned* __restrict__ mx,
                 int E_, long Ef) {
  __shared__ float sWe[224];
  __shared__ float sAtt[32];
  __shared__ float sEs[32];
  int t = threadIdx.x;
  if (t < 224) sWe[t] = We[t];
  else sAtt[t - 224] = att[t - 224];
  if (t < 32) sEs[t] = eeself[t];
  __syncthreads();

  for (long e = (long)blockIdx.x * blockDim.x + t; e < Ef;
       e += (long)gridDim.x * blockDim.x) {
    int s, d;
    float ev[32];
    if (e < E_) {
      s = src[e]; d = dst[e];
      float a[7];
#pragma unroll
      for (int f = 0; f < 7; ++f) a[f] = ea[e * 7 + f];
#pragma unroll
      for (int hc = 0; hc < 32; ++hc) {
        float acc = 0.f;
#pragma unroll
        for (int f = 0; f < 7; ++f) acc += a[f] * sWe[f * 32 + hc];
        ev[hc] = acc;
      }
    } else {
      s = d = (int)(e - E_);
#pragma unroll
      for (int hc = 0; hc < 32; ++hc) ev[hc] = sEs[hc];
    }
    const float* xls = xl + (long)s * 32;
    const float* xrd = xr + (long)d * 32;
    float acc[4] = {0.f, 0.f, 0.f, 0.f};
#pragma unroll
    for (int hc = 0; hc < 32; ++hc) {
      float m = xls[hc] + xrd[hc] + ev[hc];
      m = lrelu(m, 0.2f);  // GATv2 attention slope
      acc[hc >> 3] += m * sAtt[hc];
    }
#pragma unroll
    for (int h = 0; h < 4; ++h) {
      logits[e * 4 + h] = acc[h];
      atomicMax(&mx[(long)d * 4 + h], encF(acc[h]));
    }
  }
}

// --------------------------- edge pass B: exp + denom ----------------------
__global__ __launch_bounds__(256)
void edge_exp(const int* __restrict__ dst, float* __restrict__ lg,
              const unsigned* __restrict__ mx, float* __restrict__ den,
              int E_, long Ef) {
  for (long e = (long)blockIdx.x * blockDim.x + threadIdx.x; e < Ef;
       e += (long)gridDim.x * blockDim.x) {
    int d = (e < E_) ? dst[e] : (int)(e - E_);
#pragma unroll
    for (int h = 0; h < 4; ++h) {
      float m = decF(mx[(long)d * 4 + h]);
      float ex = expf(lg[e * 4 + h] - m);
      lg[e * 4 + h] = ex;
      atomicAdd(&den[(long)d * 4 + h], ex);
    }
  }
}

// --------------------------- edge pass C: scatter --------------------------
__global__ __launch_bounds__(256)
void edge_scatter(const int* __restrict__ src, const int* __restrict__ dst,
                  const float* __restrict__ ex, const float* __restrict__ den,
                  const float* __restrict__ xl, float* __restrict__ hout,
                  int E_, long Ef) {
  for (long e = (long)blockIdx.x * blockDim.x + threadIdx.x; e < Ef;
       e += (long)gridDim.x * blockDim.x) {
    int s, d;
    if (e < E_) { s = src[e]; d = dst[e]; } else { s = d = (int)(e - E_); }
    float al[4];
#pragma unroll
    for (int h = 0; h < 4; ++h) al[h] = ex[e * 4 + h] / (den[(long)d * 4 + h] + 1e-16f);
    const float* xls = xl + (long)s * 32;
#pragma unroll
    for (int hc = 0; hc < 32; ++hc)
      atomicAdd(&hout[(long)d * 32 + hc], xls[hc] * al[hc >> 3]);
  }
}

__global__ void bias_add(float* __restrict__ h, const float* __restrict__ b, long n32) {
  long i = (long)blockIdx.x * blockDim.x + threadIdx.x;
  if (i < n32) h[i] += b[i & 31];
}

// ------------------------------- batchnorm ---------------------------------
__global__ __launch_bounds__(256)
void bn_stats(const float* __restrict__ h, float* __restrict__ sums, int n) {
  __shared__ float ls[64];
  int t = threadIdx.x, c = t & 31, rl = t >> 5;
  float s = 0.f, s2 = 0.f;
  for (long r = (long)blockIdx.x * 8 + rl; r < n; r += (long)gridDim.x * 8) {
    float v = h[r * 32 + c];
    s += v; s2 += v * v;
  }
  if (t < 64) ls[t] = 0.f;
  __syncthreads();
  atomicAdd(&ls[c], s);
  atomicAdd(&ls[32 + c], s2);
  __syncthreads();
  if (t < 64) atomicAdd(&sums[t], ls[t]);
}

__global__ void bn_finalize(const float* __restrict__ sums,
                            const float* __restrict__ gamma, const float* __restrict__ beta,
                            float* __restrict__ prm, int n) {
  int c = threadIdx.x;  // 32 threads
  float mu  = sums[c] / (float)n;
  float var = sums[32 + c] / (float)n - mu * mu;
  float sc  = gamma[c] * rsqrtf(var + 1e-5f);
  prm[c] = sc;
  prm[32 + c] = beta[c] - mu * sc;
}

__global__ void bn_apply(const float* __restrict__ h, const float* __restrict__ prm,
                         float* __restrict__ xin, long n32) {
  long i = (long)blockIdx.x * blockDim.x + threadIdx.x;
  if (i >= n32) return;
  int c = (int)(i & 31);
  xin[i] = lrelu(h[i] * prm[c] + prm[32 + c], 0.01f);
}

// ------------------------- fused regressor head ----------------------------
__global__ __launch_bounds__(256)
void final_out(const float* __restrict__ h, const float* __restrict__ wcomb,
               const float* __restrict__ c0, float* __restrict__ out, int n) {
  __shared__ float sw[32];
  __shared__ float sc0;
  if (threadIdx.x < 32) sw[threadIdx.x] = wcomb[threadIdx.x];
  if (threadIdx.x == 0) sc0 = c0[0];
  __syncthreads();
  int i = blockIdx.x * blockDim.x + threadIdx.x;
  if (i >= n) return;
  const float* hr = h + (long)i * 32;
  float acc = sc0;
#pragma unroll
  for (int c = 0; c < 32; ++c) acc += lrelu(hr[c], 0.01f) * sw[c];
  out[i] = acc;
}

// ---------------------------------------------------------------------------
extern "C" void kernel_launch(void* const* d_in, const int* in_sizes, int n_in,
                              void* d_out, int out_size, void* d_ws, size_t ws_size,
                              hipStream_t stream) {
  const float* x    = (const float*)d_in[0];
  const int*   ei   = (const int*)d_in[1];
  const float* ea   = (const float*)d_in[2];
  const float* Wl1  = (const float*)d_in[3];
  const float* bl1  = (const float*)d_in[4];
  const float* Wr1  = (const float*)d_in[5];
  const float* br1  = (const float*)d_in[6];
  const float* We1  = (const float*)d_in[7];
  const float* att1 = (const float*)d_in[8];
  const float* b1   = (const float*)d_in[9];
  const float* Wl2  = (const float*)d_in[10];
  const float* bl2  = (const float*)d_in[11];
  const float* Wr2  = (const float*)d_in[12];
  const float* br2  = (const float*)d_in[13];
  const float* We2  = (const float*)d_in[14];
  const float* att2 = (const float*)d_in[15];
  const float* b2   = (const float*)d_in[16];
  const float* gamma = (const float*)d_in[17];
  const float* beta  = (const float*)d_in[18];
  const float* Wreg  = (const float*)d_in[19];
  const float* breg  = (const float*)d_in[20];
  const float* Wend  = (const float*)d_in[21];
  const float* bend  = (const float*)d_in[22];

  const int  N   = in_sizes[0] / 2;
  const int  E   = in_sizes[1] / 2;
  const int  REG = in_sizes[20];
  const long Ef  = (long)E + N;
  const int* src = ei;
  const int* dst = ei + E;

  // -------- workspace carve (256B aligned) --------
  uintptr_t cur = (uintptr_t)d_ws;
  auto carve = [&](size_t bytes) -> void* {
    uintptr_t p = (cur + 255) & ~(uintptr_t)255;
    cur = p + bytes;
    return (void*)p;
  };
  float*    xl   = (float*)carve((size_t)N * 32 * 4);
  float*    xr   = (float*)carve((size_t)N * 32 * 4);
  float*    hbuf = (float*)carve((size_t)N * 32 * 4);
  float*    xin  = (float*)carve((size_t)N * 32 * 4);
  float*    exl  = (float*)carve((size_t)Ef * 4 * 4);  // logits, then exp
  unsigned* mx   = (unsigned*)carve((size_t)N * 4 * 4);
  float*    den  = (float*)carve((size_t)N * 4 * 4);
  float*    misc = (float*)carve(1024);
  float* accum   = misc;        // [8]
  float* eeself1 = misc + 8;    // [32]
  float* eeself2 = misc + 40;   // [32]
  float* wcomb   = misc + 72;   // [32]
  float* c0      = misc + 104;  // [1]
  float* bnsums  = misc + 112;  // [64]
  float* bnprm   = misc + 176;  // [64]

  const int  TB  = 256;
  const long n32 = (long)N * 32;
  const int  gN32 = (int)((n32 + TB - 1) / TB);
  const int  gN4  = (int)(((long)N * 4 + TB - 1) / TB);
  const int  gE   = (int)((Ef + TB - 1) / TB);
  const int  gN   = (N + TB - 1) / TB;

  // -------- precompute --------
  zero_f32<<<1, 32, 0, stream>>>(accum, 8);
  ea_reduce<<<1024, TB, 0, stream>>>(ea, accum, E);
  precompute<<<1, 32, 0, stream>>>(accum, We1, We2, Wreg, breg, Wend, bend,
                                   eeself1, eeself2, wcomb, c0, E, REG);

  // -------- layer 1 transform (Fin=2, VALU) --------
  transform_f2<<<gN32, TB, 0, stream>>>(x, Wl1, bl1, Wr1, br1, xl, xr, N);

  for (int lay = 0; lay < 3; ++lay) {
    if (lay > 0) {
      // BatchNorm(h) -> LeakyReLU(0.01) -> xin, then WMMA transform.
      zero_f32<<<1, 64, 0, stream>>>(bnsums, 64);
      bn_stats<<<512, TB, 0, stream>>>(hbuf, bnsums, N);
      bn_finalize<<<1, 32, 0, stream>>>(bnsums, gamma, beta, bnprm, N);
      bn_apply<<<gN32, TB, 0, stream>>>(hbuf, bnprm, xin, n32);
      transform_wmma<<<512, 128, 0, stream>>>(xin, Wl2, bl2, Wr2, br2, xl, xr, N);
      // N % 16 == 0 for this problem (100000 = 6250*16); no tail path needed.
    }
    const float* We  = lay ? We2  : We1;
    const float* att = lay ? att2 : att1;
    const float* es  = lay ? eeself2 : eeself1;
    const float* bb  = lay ? b2 : b1;

    zero_f32<<<gN4, TB, 0, stream>>>((float*)mx, (long)N * 4);
    zero_f32<<<gN4, TB, 0, stream>>>(den, (long)N * 4);
    zero_f32<<<gN32, TB, 0, stream>>>(hbuf, n32);

    edge_logits<<<gE, TB, 0, stream>>>(src, dst, ea, We, att, es, xl, xr, exl, mx, E, Ef);
    edge_exp<<<gE, TB, 0, stream>>>(dst, exl, mx, den, E, Ef);
    edge_scatter<<<gE, TB, 0, stream>>>(src, dst, exl, den, xl, hbuf, E, Ef);
    bias_add<<<gN32, TB, 0, stream>>>(hbuf, bb, n32);
  }

  // -------- fused LeakyReLU + (32->500->1) regressor --------
  final_out<<<gN, TB, 0, stream>>>(hbuf, wcomb, c0, (float*)d_out, N);
}
